// MambaCompressor_82446192214775
// MI455X (gfx1250) — matrix operand order repaired and checked by reading, not compiled, not run on hardware
//
#include <hip/hip_runtime.h>
#include <hip/hip_bf16.h>
#include <stdint.h>

typedef __attribute__((ext_vector_type(16))) __bf16 v16bf;
typedef __attribute__((ext_vector_type(8)))  float  v8f;

#define Bdim 16
#define Sdim 4096
#define Hdim 1024
#define Ddim 4096
#define Kmem 512
#define Mdim (Bdim * Kmem)      /* 8192 */

#define ROWB   80               /* 64 data bytes + 16 pad: stride 20 dwords -> conflict-free */
#define ATILEA (256 * ROWB)     /* 20480 B: A tile, 256 rows */
#define ATILEB (128 * ROWB)     /* 10240 B: B tile, 128 rows */
#define BUFB   (ATILEA + ATILEB)/* 30720 B per double-buffer stage */

// ---------------- kernel 1: mem-token slot positions ----------------
__global__ void pos_kernel(const int* __restrict__ ids, const int* __restrict__ memid_p,
                           int* __restrict__ pos) {
  __shared__ int cnt[256];
  const int b = blockIdx.x, t = threadIdx.x;
  const int memid = *memid_p;
  const int* row = ids + b * Sdim;
  int local = 0;
  #pragma unroll
  for (int i = 0; i < 16; ++i) local += (row[t * 16 + i] == memid) ? 1 : 0;
  cnt[t] = local;
  __syncthreads();
  if (t == 0) {
    int run = 0;
    for (int i = 0; i < 256; ++i) { int c = cnt[i]; cnt[i] = run; run += c; }
  }
  __syncthreads();
  int slot = cnt[t];
  for (int i = 0; i < 16; ++i) {
    int s = t * 16 + i;
    if (row[s] == memid) { if (slot < Kmem) pos[b * Kmem + slot] = s; ++slot; }
  }
}

// ---------------- kernel 2: gather + f32->bf16 convert of A ----------------
struct alignas(8) bf16x4 { __bf16 a, b, c, d; };

__global__ void gather_kernel(const float* __restrict__ hidden, const int* __restrict__ pos,
                              __bf16* __restrict__ A) {
  const int m = blockIdx.x;                 // 0..8191
  const int b = m >> 9, k = m & 511;
  const int s = pos[b * Kmem + k];
  const float4* src = (const float4*)(hidden + ((size_t)b * Sdim + s) * Hdim);
  bf16x4* dst = (bf16x4*)(A + (size_t)m * Hdim);
  const int t = threadIdx.x;                // 256 threads * 4 elems = 1024
  float4 v = src[t];
  bf16x4 o{(__bf16)v.x, (__bf16)v.y, (__bf16)v.z, (__bf16)v.w};
  dst[t] = o;
}

// ---------------- kernel 3: W [H][D] f32 -> Wt [D][H] bf16 (tiled transpose) ----------------
__global__ void wtr_kernel(const float* __restrict__ W, __bf16* __restrict__ Wt) {
  __shared__ float tile[32][33];
  const int tx = threadIdx.x, ty = threadIdx.y;     // 32 x 8
  const int d0 = blockIdx.x * 32, h0 = blockIdx.y * 32;
  #pragma unroll
  for (int i = 0; i < 32; i += 8)
    tile[ty + i][tx] = W[(size_t)(h0 + ty + i) * Ddim + d0 + tx];
  __syncthreads();
  #pragma unroll
  for (int i = 0; i < 32; i += 8)
    Wt[(size_t)(d0 + ty + i) * Hdim + h0 + tx] = (__bf16)tile[tx][ty + i];
}

// ---- kernel 4: 256x128 block / 64x64 wave bf16 WMMA GEMM, async-LDS double buffer ----
__global__ __launch_bounds__(256) void gemm_kernel(
    const __bf16* __restrict__ A, const __bf16* __restrict__ Wt,
    const float* __restrict__ bias, float* __restrict__ out) {
  extern __shared__ char smem[];
  const int tid  = threadIdx.x;
  const int lane = tid & 31;
  const int wave = tid >> 5;
  const int waveM = wave & 3;               // 4 waves across M: 64 rows each
  const int waveN = wave >> 2;              // 2 waves across N: 64 cols each
  const int mBlock = blockIdx.y * 256;
  const int nBlock = blockIdx.x * 128;
  const int halfHi = (lane >> 4) & 1;       // upper half-wave -> K+8 (A and B)
  const int l16 = lane & 15;

  // ---- staging map (one row of A per thread; half a row of B per thread) ----
  // INST_OFFSET is added to BOTH the LDS and the global address, so the 16B
  // segments share one address pair and differ only in offset:0/16/32/48.
  const unsigned aGbase = (unsigned)((mBlock + tid) * Hdim * 2);          // + kb per stage
  const int      aLbase = tid * ROWB;
  const unsigned bGbase = (unsigned)((nBlock + (tid >> 1)) * Hdim * 2 + (tid & 1) * 32);
  const int      bLbase = ATILEA + (tid >> 1) * ROWB + (tid & 1) * 32;

  // accumulators seeded with bias (broadcast over M; per-lane column = l16)
  v8f acc[4][4];
  #pragma unroll
  for (int t = 0; t < 4; ++t) {
    const float bv = bias[nBlock + waveN * 64 + t * 16 + l16];
    #pragma unroll
    for (int s = 0; s < 4; ++s)
      #pragma unroll
      for (int r = 0; r < 8; ++r) acc[s][t][r] = bv;
  }

  // fragment LDS offsets within a buffer (ISA 16-bit A/B layouts)
  int aOff[4], bOff[4];
  #pragma unroll
  for (int s = 0; s < 4; ++s)
    aOff[s] = (waveM * 64 + s * 16 + l16) * ROWB + halfHi * 16;  // + {0,32}: K 0..7 / 16..23
  #pragma unroll
  for (int t = 0; t < 4; ++t)
    bOff[t] = ATILEA + (waveN * 64 + t * 16 + l16) * ROWB + halfHi * 16;  // + {0,32}

  auto stage = [&](int k0, int buf) {
    const unsigned kb = (unsigned)(k0 * 2);
    const int bo = buf * BUFB;
    const int      aL = aLbase + bo;
    const unsigned aG = aGbase + kb;
    const int      bL = bLbase + bo;
    const unsigned bG = bGbase + kb;
    asm volatile("global_load_async_to_lds_b128 %0, %1, %2 offset:0"
                 :: "v"(aL), "v"(aG), "s"((const void*)A) : "memory");
    asm volatile("global_load_async_to_lds_b128 %0, %1, %2 offset:16"
                 :: "v"(aL), "v"(aG), "s"((const void*)A) : "memory");
    asm volatile("global_load_async_to_lds_b128 %0, %1, %2 offset:32"
                 :: "v"(aL), "v"(aG), "s"((const void*)A) : "memory");
    asm volatile("global_load_async_to_lds_b128 %0, %1, %2 offset:48"
                 :: "v"(aL), "v"(aG), "s"((const void*)A) : "memory");
    asm volatile("global_load_async_to_lds_b128 %0, %1, %2 offset:0"
                 :: "v"(bL), "v"(bG), "s"((const void*)Wt) : "memory");
    asm volatile("global_load_async_to_lds_b128 %0, %1, %2 offset:16"
                 :: "v"(bL), "v"(bG), "s"((const void*)Wt) : "memory");
  };

  stage(0, 0);

  const int KT = Hdim / 32;   // 32 K-steps
  for (int kt = 0; kt < KT; ++kt) {
    const int cur = kt & 1;
    if (kt + 1 < KT) {
      stage((kt + 1) * 32, (kt + 1) & 1);
      asm volatile("s_wait_asynccnt 0x6" ::: "memory");  // current stage's 6 loads done
    } else {
      asm volatile("s_wait_asynccnt 0x0" ::: "memory");
    }
    __syncthreads();

    const char* base = smem + cur * BUFB;
    v16bf af[4], bfv[4];
    #pragma unroll
    for (int s = 0; s < 4; ++s) {
      union { int4 q[2]; v16bf v; } u;
      u.q[0] = *(const int4*)(base + aOff[s]);
      u.q[1] = *(const int4*)(base + aOff[s] + 32);
      af[s] = u.v;
    }
    #pragma unroll
    for (int t = 0; t < 4; ++t) {
      union { int4 q[2]; v16bf v; } u;
      u.q[0] = *(const int4*)(base + bOff[t]);
      u.q[1] = *(const int4*)(base + bOff[t] + 32);
      bfv[t] = u.v;
    }
    #pragma unroll
    for (int s = 0; s < 4; ++s)
      #pragma unroll
      for (int t = 0; t < 4; ++t)
        acc[s][t] = __builtin_amdgcn_wmma_f32_16x16x32_bf16(
            false, af[s], false, bfv[t], (short)0, acc[s][t], false, false);
    __syncthreads();
  }

  // ---- epilogue: single base pointer, all 128 stores use constant immediates ----
  // C/D layout: row m = r + 8*halfHi within subtile, col n = l16
  const int mg0 = mBlock + waveM * 64 + halfHi * 8;
  float* base0 = out + (size_t)mg0 * Ddim + nBlock + waveN * 64 + l16;
  #pragma unroll
  for (int s = 0; s < 4; ++s)
    #pragma unroll
    for (int r = 0; r < 8; ++r)
      #pragma unroll
      for (int t = 0; t < 4; ++t)
        base0[s * 16 * Ddim + r * Ddim + t * 16] = acc[s][t][r];
}

extern "C" void kernel_launch(void* const* d_in, const int* in_sizes, int n_in,
                              void* d_out, int out_size, void* d_ws, size_t ws_size,
                              hipStream_t stream) {
  const float* hidden = (const float*)d_in[0];
  const float* W      = (const float*)d_in[1];
  const float* bias   = (const float*)d_in[2];
  const int*   ids    = (const int*)d_in[3];
  const int*   memid  = (const int*)d_in[4];
  float* out = (float*)d_out;

  char* ws = (char*)d_ws;
  int*    pos = (int*)ws;                                              // 32 KB
  __bf16* Abf = (__bf16*)(ws + (1 << 16));                             // 16 MB
  __bf16* Wt  = (__bf16*)(ws + (1 << 16) + (size_t)Mdim * Hdim * 2);   // 8 MB

  pos_kernel   <<<Bdim, 256, 0, stream>>>(ids, memid, pos);
  gather_kernel<<<Mdim, 256, 0, stream>>>(hidden, pos, Abf);
  wtr_kernel   <<<dim3(Ddim / 32, Hdim / 32), dim3(32, 8), 0, stream>>>(W, Wt);
  gemm_kernel  <<<dim3(Ddim / 128, Mdim / 256), 256, 2 * BUFB, stream>>>(Abf, Wt, bias, out);
}
